// TransformerBlock_76871324664385
// MI455X (gfx1250) — matrix operand order repaired and checked by reading, not compiled, not run on hardware
//
#include <hip/hip_runtime.h>
#include <hip/hip_bf16.h>
#include <math.h>

// ---------------------------------------------------------------------------
// Problem constants (match reference)
// ---------------------------------------------------------------------------
#define BB   4
#define SS   1024
#define DD   1024
#define HH   16
#define DH   64          // D / H
#define THH  8
#define TDH  128         // D / TH_H
#define NT_  4
#define EE   8
#define FF   2048
#define MTOK (BB * SS)   // 4096 tokens

typedef __attribute__((ext_vector_type(16))) __bf16 v16bf;
typedef __attribute__((ext_vector_type(8)))  float  v8f;

// ---------------------------------------------------------------------------
// bf16-WMMA GEMM:  C = alpha * (A @ op(B)) + bias
//   A: (M,K) fp32 row-major. B: bT=1 -> (N,K) row-major (A@B^T); bT=0 -> (K,N).
//   REQUIRES: M % 64 == 0, N % 64 == 0, K % 32 == 0 (true for every GEMM here).
//
//   Fragments use one contiguous-K lane order (k = 16*hi + s) for both A and B of
//   v_wmma_f32_16x16x32_bf16 end-to-end (valid since A and B share it):
//     - A fragment: two contiguous 8-float global runs per lane -> straight to
//       registers (no LDS), packed-converted to bf16.
//     - B fragment: staged fragment-major in LDS (lane's 16 bf16 = one 32B
//       chunk -> 2x ds_load_b128), DOUBLE-BUFFERED: one barrier per K-step,
//       staging of tile k+1 overlaps WMMAs of tile k.
//   The 4 WMMAs share one A operand: sched_barrier groups the 8 ds_load_b128
//   ahead of the quartet (single s_wait_dscnt), and the reuse_a hint (ISA 7.12
//   OPSEL RA, legal between identical back-to-back WMMAs) caches A at the
//   matrix unit for ops 2..4.
//   Block: 128 threads (4 waves). Tile: 64x64, K-step 32.
// ---------------------------------------------------------------------------
#define TM  64
#define TN  64
#define TKK 32

__global__ __launch_bounds__(128, 3)
void gemm_bf16_kernel(const float* __restrict__ A, int lda,
                      const float* __restrict__ Bm, int ldb, int bT,
                      const float* __restrict__ bias,
                      float* __restrict__ C, int ldc,
                      int M, int N, int K, float alpha)
{
    // [buf][nt][lane][frag elem]  -- 2 x 4KB
    __shared__ __attribute__((aligned(32))) __bf16 sBf[2][4][32][16];

    const int tid  = threadIdx.x;
    const int lane = tid & 31;
    const int wave = tid >> 5;            // 0..3 -> wave's 16-row strip
    const int m0   = blockIdx.y * TM;
    const int n0   = blockIdx.x * TN;
    const int r16  = lane & 15;
    const int hi   = lane >> 4;           // lane-half select

    // A fragment addressing (lane hi=0 -> K {0..15}; hi=1 -> K {16..31})
    const float* arow  = A + (size_t)(m0 + wave * 16 + r16) * lda;
    const int    koffA = hi * 16;

    // bT=1 staging: thread owns fragment (nt = wave, lane)
    const int ntS = wave;
    const int nS  = n0 + ntS * 16 + r16;  // global column staged by this thread
    const int kS  = hi * 16;              // K offset within tile (B layout)

    // bT=0 staging: thread owns one K-row x 16-col chunk
    const int kR  = tid >> 2;             // local k row 0..31
    const int nCh = tid & 3;              // 16-wide N chunk (== nt)
    const int hiR = kR >> 4;
    const int iR  = kR & 15;              // fragment element index

    auto stageB = [&](int k0, int buf) {
        float tv[16];
        if (bT) {
            const float4* p = (const float4*)(Bm + (size_t)nS * ldb + k0 + kS);
            #pragma unroll
            for (int i = 0; i < 4; ++i) {
                const float4 v = p[i];
                tv[4 * i + 0] = v.x; tv[4 * i + 1] = v.y;
                tv[4 * i + 2] = v.z; tv[4 * i + 3] = v.w;
            }
            __bf16* dst = &sBf[buf][ntS][lane][0];
            #pragma unroll
            for (int i = 0; i < 16; ++i) dst[i] = (__bf16)tv[i];
        } else {
            const float4* p = (const float4*)(Bm + (size_t)(k0 + kR) * ldb
                                              + n0 + nCh * 16);
            #pragma unroll
            for (int i = 0; i < 4; ++i) {
                const float4 v = p[i];
                tv[4 * i + 0] = v.x; tv[4 * i + 1] = v.y;
                tv[4 * i + 2] = v.z; tv[4 * i + 3] = v.w;
            }
            #pragma unroll
            for (int j = 0; j < 16; ++j)
                sBf[buf][nCh][hiR * 16 + j][iR] = (__bf16)tv[j];
        }
    };

    v8f acc[4] = {};

    stageB(0, 0);
    int pb = 0;
    for (int k0 = 0; k0 < K; k0 += TKK, pb ^= 1) {
        __syncthreads();   // staging of buffer pb complete; old buffer consumed

        // ---- A fragment straight from global into registers (2x 32B runs) ----
        v16bf afrag;
        {
            const float4* pa = (const float4*)(arow + k0 + koffA);
            const float4* pc = (const float4*)(arow + k0 + 8 + koffA);
            const float4 a0 = pa[0], a1 = pa[1];
            const float4 a2 = pc[0], a3 = pc[1];
            float av[16];
            av[0]  = a0.x; av[1]  = a0.y; av[2]  = a0.z; av[3]  = a0.w;
            av[4]  = a1.x; av[5]  = a1.y; av[6]  = a1.z; av[7]  = a1.w;
            av[8]  = a2.x; av[9]  = a2.y; av[10] = a2.z; av[11] = a2.w;
            av[12] = a3.x; av[13] = a3.y; av[14] = a3.z; av[15] = a3.w;
            #pragma unroll
            for (int i = 0; i < 16; ++i) afrag[i] = (__bf16)av[i];
        }

        // ---- overlap: stage next K-tile into the other buffer ----
        if (k0 + TKK < K) {
            stageB(k0 + TKK, pb ^ 1);
            __builtin_prefetch(arow + k0 + 2 * TKK + koffA, 0, 1);  // 2 tiles out
        }

        // ---- issue all 8 ds_load_b128 first, then 4 WMMAs back-to-back ----
        const v16bf b0 = *(const v16bf*)&sBf[pb][0][lane][0];
        const v16bf b1 = *(const v16bf*)&sBf[pb][1][lane][0];
        const v16bf b2 = *(const v16bf*)&sBf[pb][2][lane][0];
        const v16bf b3 = *(const v16bf*)&sBf[pb][3][lane][0];
        __builtin_amdgcn_sched_barrier(0);   // keep loads grouped before WMMAs

        acc[0] = __builtin_amdgcn_wmma_f32_16x16x32_bf16(
            false, afrag, false, b0, (short)0, acc[0], false, false);
        // identical opcode + identical A registers -> A-reuse hint is legal
        acc[1] = __builtin_amdgcn_wmma_f32_16x16x32_bf16(
            false, afrag, false, b1, (short)0, acc[1], true, false);
        acc[2] = __builtin_amdgcn_wmma_f32_16x16x32_bf16(
            false, afrag, false, b2, (short)0, acc[2], true, false);
        acc[3] = __builtin_amdgcn_wmma_f32_16x16x32_bf16(
            false, afrag, false, b3, (short)0, acc[3], true, false);
    }

    // ---- writeback: acc[nt][j] -> row m0+wave*16+hi*8+j, col n0+nt*16+r16 ----
    #pragma unroll
    for (int nt = 0; nt < 4; ++nt) {
        const int col = n0 + nt * 16 + r16;
        const float bv = bias ? bias[col] : 0.f;
        #pragma unroll
        for (int j = 0; j < 8; ++j) {
            const int row = m0 + wave * 16 + hi * 8 + j;
            C[(size_t)row * ldc + col] = acc[nt][j] * alpha + bv;
        }
    }
}

// ---------------------------------------------------------------------------
// RoPE: xr = rot(x). FRAC=1.0 -> apply_dim = D, half = 512, 256 base freqs
// duplicated. One thread per (token, pair).
// ---------------------------------------------------------------------------
__global__ void rope_kernel(const float* __restrict__ x, float* __restrict__ xr,
                            int total)
{
    int idx = blockIdx.x * blockDim.x + threadIdx.x;
    if (idx >= total) return;
    const int t  = idx >> 9;          // token  (total = tokens * 512)
    const int j  = idx & 511;         // pair index
    const int s  = t & (SS - 1);      // position within sequence
    const int jm = j & 255;
    const float ang = (float)s * __expf(-logf(10000.f) * (2.f * (float)jm) / 512.f);
    const float c  = cosf(ang);
    const float sn = sinf(ang);
    const size_t base = (size_t)t * DD + 2 * j;
    const float a  = x[base];
    const float b2 = x[base + 1];
    xr[base]     = a * c  - b2 * sn;
    xr[base + 1] = a * sn + b2 * c;
}

// ---------------------------------------------------------------------------
// out = LayerNorm(a + r) * g + b  (row length D=1024, one block per token)
// ---------------------------------------------------------------------------
__global__ __launch_bounds__(256)
void add_ln_kernel(const float* __restrict__ a, const float* __restrict__ r,
                   const float* __restrict__ g, const float* __restrict__ be,
                   float* __restrict__ out)
{
    __shared__ float red[256];
    const int t   = blockIdx.x;
    const int tid = threadIdx.x;
    const float* pa = a + (size_t)t * DD;
    const float* pr = r + (size_t)t * DD;

    float v[4];
    float s = 0.f;
    #pragma unroll
    for (int i = 0; i < 4; ++i) {
        const int d = tid + i * 256;
        v[i] = pa[d] + pr[d];
        s += v[i];
    }
    red[tid] = s; __syncthreads();
    for (int o = 128; o > 0; o >>= 1) {
        if (tid < o) red[tid] += red[tid + o];
        __syncthreads();
    }
    const float mean = red[0] * (1.f / DD);
    __syncthreads();

    s = 0.f;
    #pragma unroll
    for (int i = 0; i < 4; ++i) { const float d0 = v[i] - mean; s += d0 * d0; }
    red[tid] = s; __syncthreads();
    for (int o = 128; o > 0; o >>= 1) {
        if (tid < o) red[tid] += red[tid + o];
        __syncthreads();
    }
    const float inv = rsqrtf(red[0] * (1.f / DD) + 1e-5f);

    #pragma unroll
    for (int i = 0; i < 4; ++i) {
        const int d = tid + i * 256;
        out[(size_t)t * DD + d] = (v[i] - mean) * inv * g[d] + be[d];
    }
}

// ---------------------------------------------------------------------------
// In-place row softmax, one block per row of length L
// ---------------------------------------------------------------------------
__global__ __launch_bounds__(256)
void softmax_rows_kernel(float* __restrict__ p, int L)
{
    __shared__ float red[256];
    float* row = p + (size_t)blockIdx.x * L;
    const int tid = threadIdx.x;

    float m = -3.4e38f;
    for (int i = tid; i < L; i += 256) m = fmaxf(m, row[i]);
    red[tid] = m; __syncthreads();
    for (int o = 128; o > 0; o >>= 1) {
        if (tid < o) red[tid] = fmaxf(red[tid], red[tid + o]);
        __syncthreads();
    }
    m = red[0]; __syncthreads();

    float s = 0.f;
    for (int i = tid; i < L; i += 256) {
        const float e = __expf(row[i] - m);
        row[i] = e;
        s += e;
    }
    red[tid] = s; __syncthreads();
    for (int o = 128; o > 0; o >>= 1) {
        if (tid < o) red[tid] += red[tid + o];
        __syncthreads();
    }
    const float inv = 1.f / red[0];
    for (int i = tid; i < L; i += 256) row[i] *= inv;
}

// ---------------------------------------------------------------------------
// avg[t][d] = mean over NT of thoughts[t][nt*D + d]
// ---------------------------------------------------------------------------
__global__ void avg_thoughts_kernel(const float* __restrict__ th,
                                    float* __restrict__ avg, int total)
{
    int idx = blockIdx.x * blockDim.x + threadIdx.x;
    if (idx >= total) return;
    const int t = idx >> 10;
    const int d = idx & (DD - 1);
    const size_t base = (size_t)t * (NT_ * DD) + d;
    avg[idx] = 0.25f * (th[base] + th[base + DD] + th[base + 2 * DD] + th[base + 3 * DD]);
}

// h <- h * (h * sigmoid(h))
__global__ void swiglu_kernel(float* __restrict__ h, int n)
{
    int idx = blockIdx.x * blockDim.x + threadIdx.x;
    if (idx >= n) return;
    const float v = h[idx];
    const float sg = 1.f / (1.f + __expf(-v));
    h[idx] = v * v * sg;
}

// ---------------------------------------------------------------------------
// Top-2 gating: comb[t][e] = normalized top-2 softmax weights (dense E-vec)
// Block: 256 threads = 8 waves; wave e computes dot(x[t], gate_W[e])
// ---------------------------------------------------------------------------
__global__ __launch_bounds__(256)
void gate_topk_kernel(const float* __restrict__ x, const float* __restrict__ W,
                      float* __restrict__ comb)
{
    __shared__ float dots[EE];
    const int t    = blockIdx.x;
    const int wave = threadIdx.x >> 5;
    const int lane = threadIdx.x & 31;
    const float* px = x + (size_t)t * DD;
    const float* pw = W + (size_t)wave * DD;
    float s = 0.f;
    for (int i = lane; i < DD; i += 32) s += px[i] * pw[i];
    #pragma unroll
    for (int o = 16; o > 0; o >>= 1) s += __shfl_xor(s, o, 32);
    if (lane == 0) dots[wave] = s;
    __syncthreads();
    if (threadIdx.x == 0) {
        float p[EE];
        float mx = -3.4e38f;
        for (int e = 0; e < EE; ++e) { p[e] = dots[e]; mx = fmaxf(mx, p[e]); }
        float sum = 0.f;
        for (int e = 0; e < EE; ++e) { p[e] = __expf(p[e] - mx); sum += p[e]; }
        const float inv = 1.f / sum;
        for (int e = 0; e < EE; ++e) p[e] *= inv;
        int i0 = 0;
        for (int e = 1; e < EE; ++e) if (p[e] > p[i0]) i0 = e;
        int i1 = (i0 == 0) ? 1 : 0;
        for (int e = 0; e < EE; ++e) if (e != i0 && p[e] > p[i1]) i1 = e;
        const float gs = 1.f / (p[i0] + p[i1]);
        for (int e = 0; e < EE; ++e) comb[(size_t)t * EE + e] = 0.f;
        comb[(size_t)t * EE + i0] = p[i0] * gs;
        comb[(size_t)t * EE + i1] = p[i1] * gs;
    }
}

// coef[t][0..1] = softmax(x[t] @ coef_W^T + coef_b)  (block = 64 thr = 2 waves)
__global__ __launch_bounds__(64)
void coef_kernel(const float* __restrict__ x, const float* __restrict__ W,
                 const float* __restrict__ bc, float* __restrict__ coef)
{
    __shared__ float dots[2];
    const int t    = blockIdx.x;
    const int wave = threadIdx.x >> 5;
    const int lane = threadIdx.x & 31;
    const float* px = x + (size_t)t * DD;
    const float* pw = W + (size_t)wave * DD;
    float s = 0.f;
    for (int i = lane; i < DD; i += 32) s += px[i] * pw[i];
    #pragma unroll
    for (int o = 16; o > 0; o >>= 1) s += __shfl_xor(s, o, 32);
    if (lane == 0) dots[wave] = s;
    __syncthreads();
    if (threadIdx.x == 0) {
        const float a0 = dots[0] + bc[0];
        const float a1 = dots[1] + bc[1];
        const float m  = fmaxf(a0, a1);
        const float e0 = __expf(a0 - m);
        const float e1 = __expf(a1 - m);
        const float inv = 1.f / (e0 + e1);
        coef[(size_t)t * 2 + 0] = e0 * inv;
        coef[(size_t)t * 2 + 1] = e1 * inv;
    }
}

// acc[t][d] (+)= comb[t][e] * y[t][d]
__global__ void moe_accum_kernel(const float* __restrict__ y,
                                 const float* __restrict__ comb, int e,
                                 float* __restrict__ acc, int init, int n)
{
    int idx = blockIdx.x * blockDim.x + threadIdx.x;
    if (idx >= n) return;
    const int t = idx >> 10;
    const float w = comb[(size_t)t * EE + e];
    const float v = w * y[idx];
    acc[idx] = init ? v : (acc[idx] + v);
}

// ffn = acc * coef[:,0] + yr * coef[:,1]
__global__ void combine_kernel(const float* __restrict__ acc,
                               const float* __restrict__ yr,
                               const float* __restrict__ coef,
                               float* __restrict__ ffn, int n)
{
    int idx = blockIdx.x * blockDim.x + threadIdx.x;
    if (idx >= n) return;
    const int t = idx >> 10;
    ffn[idx] = acc[idx] * coef[(size_t)t * 2] + yr[idx] * coef[(size_t)t * 2 + 1];
}

// ---------------------------------------------------------------------------
// Host orchestration
// ---------------------------------------------------------------------------
extern "C" void kernel_launch(void* const* d_in, const int* in_sizes, int n_in,
                              void* d_out, int out_size, void* d_ws, size_t ws_size,
                              hipStream_t stream)
{
    (void)in_sizes; (void)n_in; (void)out_size; (void)ws_size;

    const float* x       = (const float*)d_in[0];
    const float* attn_Wi = (const float*)d_in[1];
    const float* attn_bi = (const float*)d_in[2];
    const float* attn_Wo = (const float*)d_in[3];
    const float* attn_bo = (const float*)d_in[4];
    const float* th_Wg   = (const float*)d_in[5];
    const float* th_bg   = (const float*)d_in[6];
    const float* th_Wi   = (const float*)d_in[7];
    const float* th_bi   = (const float*)d_in[8];
    const float* th_Wo   = (const float*)d_in[9];
    const float* th_bo   = (const float*)d_in[10];
    const float* gate_W  = (const float*)d_in[11];
    const float* exp_W1  = (const float*)d_in[12];
    const float* exp_b1  = (const float*)d_in[13];
    const float* exp_W2  = (const float*)d_in[14];
    const float* exp_b2  = (const float*)d_in[15];
    const float* res_W1  = (const float*)d_in[16];
    const float* res_b1  = (const float*)d_in[17];
    const float* res_W2  = (const float*)d_in[18];
    const float* res_b2  = (const float*)d_in[19];
    const float* coef_W  = (const float*)d_in[20];
    const float* coef_b  = (const float*)d_in[21];
    const float* ln1_g   = (const float*)d_in[22];
    const float* ln1_b   = (const float*)d_in[23];
    const float* ln2_g   = (const float*)d_in[24];
    const float* ln2_b   = (const float*)d_in[25];
    const float* ln3_g   = (const float*)d_in[26];
    const float* ln3_b   = (const float*)d_in[27];
    float* out = (float*)d_out;

    // ---- workspace layout (floats) ----
    float* w = (float*)d_ws;
    size_t off = 0;
    float* f_xr   = w + off; off += (size_t)MTOK * DD;        // rotated x
    float* f_q    = w + off; off += (size_t)MTOK * DD;
    float* f_k    = w + off; off += (size_t)MTOK * DD;
    float* f_v    = w + off; off += (size_t)MTOK * DD;
    float* f_sc   = w + off; off += (size_t)SS * SS;          // one head's scores
    float* f_o    = w + off; off += (size_t)MTOK * DD;        // attn context
    float* f_ao   = w + off; off += (size_t)MTOK * DD;        // proj out
    float* f_x1   = w + off; off += (size_t)MTOK * DD;
    float* f_th   = w + off; off += (size_t)MTOK * NT_ * DD;  // thoughts / ffn
    float* f_avg  = w + off; off += (size_t)MTOK * DD;
    float* f_x2   = w + off; off += (size_t)MTOK * DD;
    float* f_h    = w + off; off += (size_t)MTOK * FF;        // expert hidden
    float* f_y    = w + off; off += (size_t)MTOK * DD;        // expert out / yr
    float* f_acc  = w + off; off += (size_t)MTOK * DD;
    float* f_comb = w + off; off += (size_t)MTOK * EE;
    float* f_coef = w + off; off += (size_t)MTOK * 2;

    auto gemm = [&](const float* A, int lda, const float* Bm, int ldb, int bT,
                    const float* bias, float* C, int ldc,
                    int M, int N, int K, float alpha) {
        dim3 grid((N + TN - 1) / TN, (M + TM - 1) / TM);
        gemm_bf16_kernel<<<grid, 128, 0, stream>>>(A, lda, Bm, ldb, bT, bias,
                                                   C, ldc, M, N, K, alpha);
    };

    const int nElem = MTOK * DD;               // 4M
    const int eThr  = 256;
    const int eBlkD = (nElem + eThr - 1) / eThr;

    // ---- 1. RoPE (q and k are both rot(x)) ----
    {
        const int total = MTOK * 512;
        rope_kernel<<<(total + eThr - 1) / eThr, eThr, 0, stream>>>(x, f_xr, total);
    }

    // ---- 2. Attention QKV projections ----
    gemm(f_xr, DD, attn_Wi,                       DD, 1, attn_bi,          f_q, DD, MTOK, DD, DD, 1.f);
    gemm(f_xr, DD, attn_Wi + (size_t)DD * DD,     DD, 1, attn_bi + DD,     f_k, DD, MTOK, DD, DD, 1.f);
    gemm(x,    DD, attn_Wi + (size_t)2 * DD * DD, DD, 1, attn_bi + 2 * DD, f_v, DD, MTOK, DD, DD, 1.f);

    // ---- 3. Per-(batch, head) attention: scores -> softmax -> context ----
    const float scale1 = 0.125f;               // 1/sqrt(64)
    for (int b = 0; b < BB; ++b) {
        for (int h = 0; h < HH; ++h) {
            const float* qh = f_q + (size_t)b * SS * DD + h * DH;
            const float* kh = f_k + (size_t)b * SS * DD + h * DH;
            const float* vh = f_v + (size_t)b * SS * DD + h * DH;
            float* oh = f_o + (size_t)b * SS * DD + h * DH;
            gemm(qh, DD, kh, DD, 1, nullptr, f_sc, SS, SS, SS, DH, scale1);
            softmax_rows_kernel<<<SS, 256, 0, stream>>>(f_sc, SS);
            gemm(f_sc, SS, vh, DD, 0, nullptr, oh, DD, SS, DH, SS, 1.f);
        }
    }

    // ---- 4. Output projection + LN1 ----
    gemm(f_o, DD, attn_Wo, DD, 1, attn_bo, f_ao, DD, MTOK, DD, DD, 1.f);
    add_ln_kernel<<<MTOK, 256, 0, stream>>>(x, f_ao, ln1_g, ln1_b, f_x1);

    // ---- 5. Thought generation + average ----
    gemm(f_x1, DD, th_Wg, DD, 1, th_bg, f_th, NT_ * DD, MTOK, NT_ * DD, DD, 1.f);
    avg_thoughts_kernel<<<eBlkD, eThr, 0, stream>>>(f_th, f_avg, nElem);

    // ---- 6. Thought MHA: q from avg, k/v from x1 ----
    gemm(f_avg, DD, th_Wi,                       DD, 1, th_bi,          f_q, DD, MTOK, DD, DD, 1.f);
    gemm(f_x1,  DD, th_Wi + (size_t)DD * DD,     DD, 1, th_bi + DD,     f_k, DD, MTOK, DD, DD, 1.f);
    gemm(f_x1,  DD, th_Wi + (size_t)2 * DD * DD, DD, 1, th_bi + 2 * DD, f_v, DD, MTOK, DD, DD, 1.f);

    const float scale2 = 0.08838834764831845f; // 1/sqrt(128)
    for (int b = 0; b < BB; ++b) {
        for (int h = 0; h < THH; ++h) {
            const float* qh = f_q + (size_t)b * SS * DD + h * TDH;
            const float* kh = f_k + (size_t)b * SS * DD + h * TDH;
            const float* vh = f_v + (size_t)b * SS * DD + h * TDH;
            float* oh = f_o + (size_t)b * SS * DD + h * TDH;
            gemm(qh, DD, kh, DD, 1, nullptr, f_sc, SS, SS, SS, TDH, scale2);
            softmax_rows_kernel<<<SS, 256, 0, stream>>>(f_sc, SS);
            gemm(f_sc, SS, vh, DD, 0, nullptr, oh, DD, SS, TDH, SS, 1.f);
        }
    }
    gemm(f_o, DD, th_Wo, DD, 1, th_bo, f_ao, DD, MTOK, DD, DD, 1.f);
    add_ln_kernel<<<MTOK, 256, 0, stream>>>(f_x1, f_ao, ln3_g, ln3_b, f_x2);

    // ---- 7. MoE: gating + dense per-expert FFN (reference runs all experts) ----
    gate_topk_kernel<<<MTOK, 256, 0, stream>>>(f_x2, gate_W, f_comb);
    const int nH   = MTOK * FF;
    const int hBlk = (nH + eThr - 1) / eThr;
    for (int e = 0; e < EE; ++e) {
        gemm(f_x2, DD, exp_W1 + (size_t)e * FF * DD, DD, 1, exp_b1 + (size_t)e * FF,
             f_h, FF, MTOK, FF, DD, 1.f);
        swiglu_kernel<<<hBlk, eThr, 0, stream>>>(f_h, nH);
        gemm(f_h, FF, exp_W2 + (size_t)e * DD * FF, FF, 1, exp_b2 + (size_t)e * DD,
             f_y, DD, MTOK, DD, FF, 1.f);
        moe_accum_kernel<<<eBlkD, eThr, 0, stream>>>(f_y, f_comb, e, f_acc,
                                                     (e == 0) ? 1 : 0, nElem);
    }

    // ---- 8. Residual FFN path ----
    gemm(f_x2, DD, res_W1, DD, 1, res_b1, f_h, FF, MTOK, FF, DD, 1.f);
    swiglu_kernel<<<hBlk, eThr, 0, stream>>>(f_h, nH);
    gemm(f_h, FF, res_W2, FF, 1, res_b2, f_y, DD, MTOK, DD, FF, 1.f);

    // ---- 9. Mixture coefficients + combine + final LN ----
    coef_kernel<<<MTOK, 64, 0, stream>>>(f_x2, coef_W, coef_b, f_coef);
    combine_kernel<<<eBlkD, eThr, 0, stream>>>(f_acc, f_y, f_coef, f_th, nElem);
    add_ln_kernel<<<MTOK, 256, 0, stream>>>(f_x2, f_th, ln2_g, ln2_b, out);
}
